// MolecularGNN_9964324126753
// MI455X (gfx1250) — matrix operand order repaired and checked by reading, not compiled, not run on hardware
//
#include <hip/hip_runtime.h>
#include <hip/hip_bf16.h>

#define NN   100000
#define EE   1600000
#define HH   128
#define GG   2048
#define KK   5
#define BN_EPS 1e-5f

typedef __attribute__((ext_vector_type(16))) _Float16 v16h;
typedef __attribute__((ext_vector_type(8)))  _Float16 v8h;
typedef __attribute__((ext_vector_type(8)))  float    v8f;

// ---------------- elementwise helpers ----------------
__global__ void k_fill1(float* p, int n) {
    int i = blockIdx.x * blockDim.x + threadIdx.x;
    if (i < n) p[i] = 1.0f;
}
__global__ void k_zero(float* p, int n) {
    int i = blockIdx.x * blockDim.x + threadIdx.x;
    if (i < n) p[i] = 0.0f;
}
__global__ void k_deg_acc(const int* __restrict__ dstv, float* __restrict__ deg, int e) {
    int i = blockIdx.x * blockDim.x + threadIdx.x;
    if (i < e) atomicAdd(&deg[dstv[i]], 1.0f);
}
__global__ void k_rsqrt(const float* __restrict__ a, float* __restrict__ b, int n) {
    int i = blockIdx.x * blockDim.x + threadIdx.x;
    if (i < n) b[i] = rsqrtf(a[i]);
}
__global__ void k_f2h(const float* __restrict__ a, _Float16* __restrict__ b, int n) {
    int i = blockIdx.x * blockDim.x + threadIdx.x;
    if (i < n) b[i] = (_Float16)a[i];
}
// W is [K=128][N=128] f32 (fi x fo). Produce Wt[n][k] f16 for contiguous B-fragment loads.
__global__ void k_wt(const float* __restrict__ W, _Float16* __restrict__ Wt) {
    int i = blockIdx.x * blockDim.x + threadIdx.x;
    if (i < HH * HH) {
        int n = i >> 7, k = i & 127;
        Wt[i] = (_Float16)W[k * HH + n];
    }
}

// ---------------- WMMA GEMM: hW[N][128] = h16[N][128] @ W[128][128] ----------------
// 256 threads = 8 waves; each wave computes a 16-row strip across all 128 output
// columns: 8 tiles x 4 k-steps = 32 v_wmma_f32_16x16x32_f16 per wave.
// Wt is staged in LDS; A fragments preloaded (overlapping the LDS fill);
// B fragments double-buffered from LDS across the unrolled WMMA loop.
__global__ __launch_bounds__(256) void k_gemm(const _Float16* __restrict__ h16,
                                              const _Float16* __restrict__ Wt,
                                              float* __restrict__ out, int nrows) {
    __shared__ _Float16 wlds[HH * HH];          // 32 KB of 320 KB WGP LDS

    // cooperative LDS fill: 2048 x uint4 = 32 KB
    {
        const uint4* s = (const uint4*)Wt;
        uint4* d = (uint4*)wlds;
        #pragma unroll
        for (int i = 0; i < 8; ++i)
            d[threadIdx.x + i * 256] = s[threadIdx.x + i * 256];
    }

    const int wave = threadIdx.x >> 5;
    const int lane = threadIdx.x & 31;
    const int r0 = (blockIdx.x * 8 + wave) * 16;
    const int half = (lane >> 4) & 1;           // 0: lanes 0-15, 1: lanes 16-31
    const int l15  = lane & 15;

    // Preload all 4 A fragments now so the global loads overlap the LDS fill.
    // Row clamped so speculative loads for tail waves stay in-bounds.
    const int mrow = min(r0 + l15, nrows - 1);
    const _Float16* arow = h16 + (size_t)mrow * HH;
    v16h afrag[4];
    #pragma unroll
    for (int kb = 0; kb < 4; ++kb) {
        v8h alo = *(const v8h*)(arow + kb * 32 + half * 16);
        v8h ahi = *(const v8h*)(arow + kb * 32 + half * 16 + 8);
        #pragma unroll
        for (int i = 0; i < 8; ++i) { afrag[kb][i] = alo[i]; afrag[kb][i + 8] = ahi[i]; }
    }

    __syncthreads();
    if (r0 >= nrows) return;                    // wave-uniform: EXEC all-ones for WMMA

    auto ldb = [&](int f) -> v16h {             // B fragment f = kb*8 + nt from LDS
        int kb = f >> 3, nt = f & 7;
        return *(const v16h*)(wlds + (nt * 16 + l15) * HH + kb * 32 + half * 16);
    };

    v8f acc[8] = {};
    v16h bcur = ldb(0);
    #pragma unroll
    for (int f = 0; f < 32; ++f) {              // double-buffered: load f+1, mma f
        v16h bnext = (f < 31) ? ldb(f + 1) : bcur;
        acc[f & 7] = __builtin_amdgcn_wmma_f32_16x16x32_f16(
            false, afrag[f >> 3], false, bcur, (short)0, acc[f & 7], false, false);
        bcur = bnext;
    }

    // D layout: VGPR r -> row (r0 + half*8 + r), col (nt*16 + lane&15)
    const int mbase = r0 + half * 8;
    #pragma unroll
    for (int nt = 0; nt < 8; ++nt) {
        #pragma unroll
        for (int r = 0; r < 8; ++r)
            out[(size_t)(mbase + r) * HH + nt * 16 + l15] = acc[nt][r];
    }
}

// ---------------- GCN aggregation ----------------
// agg = hW * snorm + bias   (snorm = dinv^2, bias broadcast per channel)
__global__ void k_agg_init(const float* __restrict__ hW, const float* __restrict__ dinv,
                           const float* __restrict__ bias, float* __restrict__ agg, int n) {
    int i = blockIdx.x * blockDim.x + threadIdx.x;
    if (i < n) {
        int row = i >> 7, c = i & 127;
        float dv = dinv[row];
        agg[i] = hW[i] * dv * dv + bias[c];
    }
}
// One wave per edge: lane t handles 4 floats (float4 gather, 4x f32 atomics; both
// buffers are L2-resident at 51.2 MB each).
__global__ __launch_bounds__(256) void k_scatter(const float* __restrict__ hW,
                                                 const int* __restrict__ srcv,
                                                 const int* __restrict__ dstv,
                                                 const float* __restrict__ dinv,
                                                 float* __restrict__ agg) {
    int e = blockIdx.x * 8 + (threadIdx.x >> 5);
    int lane = threadIdx.x & 31;
    if (e >= EE) return;
    int s = srcv[e], d = dstv[e];
    float w = dinv[s] * dinv[d];
    const float4 v = *(const float4*)(hW + (size_t)s * HH + lane * 4);
    float* ap = agg + (size_t)d * HH + lane * 4;
    atomicAdd(ap + 0, v.x * w);
    atomicAdd(ap + 1, v.y * w);
    atomicAdd(ap + 2, v.z * w);
    atomicAdd(ap + 3, v.w * w);
}

// ---------------- BatchNorm (training stats, biased variance) ----------------
__global__ __launch_bounds__(128) void k_bn_stats(const float* __restrict__ agg,
                                                  float* __restrict__ stats) {
    int c = threadIdx.x;
    float s = 0.f, q = 0.f;
    for (int r = blockIdx.x; r < NN; r += gridDim.x) {
        float v = agg[(size_t)r * HH + c];
        s += v; q += v * v;
    }
    atomicAdd(&stats[c], s);
    atomicAdd(&stats[HH + c], q);
}
__global__ void k_bn_fin(const float* __restrict__ stats, const float* __restrict__ gamma,
                         const float* __restrict__ beta, float* __restrict__ ss) {
    int c = threadIdx.x;
    float mu  = stats[c] * (1.0f / NN);
    float var = stats[HH + c] * (1.0f / NN) - mu * mu;
    float inv = rsqrtf(var + BN_EPS);
    float sc = gamma[c] * inv;
    ss[c] = sc;
    ss[HH + c] = beta[c] - mu * sc;
}
// h = relu(agg*scale + shift); in-place f32 + f16 copy for the next GEMM
__global__ void k_bn_apply(float* __restrict__ agg, const float* __restrict__ ss,
                           _Float16* __restrict__ h16, int n) {
    int i = blockIdx.x * blockDim.x + threadIdx.x;
    if (i < n) {
        int c = i & 127;
        float v = fmaxf(agg[i] * ss[c] + ss[HH + c], 0.0f);
        agg[i] = v;
        h16[i] = (_Float16)v;
    }
}

// ---------------- global mean pool ----------------
__global__ void k_pool(const float* __restrict__ h, const int* __restrict__ batch,
                       float* __restrict__ pooled, float* __restrict__ counts, int n) {
    int i = blockIdx.x * blockDim.x + threadIdx.x;
    if (i < n) {
        int row = i >> 7, c = i & 127;
        int g = batch[row];
        atomicAdd(&pooled[(size_t)g * HH + c], h[i]);
        if (c == 0) atomicAdd(&counts[g], 1.0f);
    }
}

// ---------------- 5 MLP heads: Linear(128,64) -> ReLU -> Linear(64,1) ----------------
__global__ void k_heads(const float* __restrict__ pooled, const float* __restrict__ counts,
                        const float* __restrict__ HW1, const float* __restrict__ Hb1,
                        const float* __restrict__ HW2, const float* __restrict__ Hb2,
                        float* __restrict__ out) {
    int t = blockIdx.x * blockDim.x + threadIdx.x;
    if (t >= KK * GG) return;
    int k = t / GG, g = t % GG;
    float inv = 1.0f / fmaxf(counts[g], 1.0f);
    float z[64];
    #pragma unroll
    for (int m = 0; m < 64; ++m) z[m] = Hb1[k * 64 + m];
    const float* pg = pooled + (size_t)g * HH;
    const float* w1 = HW1 + (size_t)k * HH * 64;
    for (int h = 0; h < HH; ++h) {
        float ph = pg[h] * inv;
        const float* w1h = w1 + h * 64;
        #pragma unroll
        for (int m = 0; m < 64; ++m) z[m] = fmaf(ph, w1h[m], z[m]);
    }
    float acc = Hb2[k];
    const float* w2 = HW2 + k * 64;
    #pragma unroll
    for (int m = 0; m < 64; ++m) acc += fmaxf(z[m], 0.0f) * w2[m];
    out[k * GG + g] = acc;
}

// ---------------- driver ----------------
extern "C" void kernel_launch(void* const* d_in, const int* in_sizes, int n_in,
                              void* d_out, int out_size, void* d_ws, size_t ws_size,
                              hipStream_t stream) {
    (void)in_sizes; (void)n_in; (void)out_size; (void)ws_size;
    const float* x     = (const float*)d_in[0];
    const int*   ei    = (const int*)d_in[1];
    const int*   srcv  = ei;
    const int*   dstv  = ei + EE;
    const int*   batch = (const int*)d_in[3];
    const float* W0    = (const float*)d_in[4];
    const float* Wrest = (const float*)d_in[5];
    const float* bvec  = (const float*)d_in[6];
    const float* gamma = (const float*)d_in[7];
    const float* beta  = (const float*)d_in[8];
    const float* HW1   = (const float*)d_in[9];
    const float* Hb1   = (const float*)d_in[10];
    const float* HW2   = (const float*)d_in[11];
    const float* Hb2   = (const float*)d_in[12];
    float* out = (float*)d_out;

    char* ws = (char*)d_ws;
    size_t o = 0;
    auto take = [&](size_t bytes) -> char* {
        char* p = ws + o;
        o = (o + bytes + 255) & ~(size_t)255;
        return p;
    };
    _Float16* h16   = (_Float16*)take((size_t)NN * HH * 2);
    float*    hW    = (float*)   take((size_t)NN * HH * 4);
    float*    agg   = (float*)   take((size_t)NN * HH * 4);   // doubles as h32 after BN
    float*    deg   = (float*)   take((size_t)NN * 4);
    float*    dinv  = (float*)   take((size_t)NN * 4);
    _Float16* Wt    = (_Float16*)take((size_t)HH * HH * 2);
    float*    stats = (float*)   take((size_t)2 * HH * 4);
    float*    ss    = (float*)   take((size_t)2 * HH * 4);
    float*    pooled= (float*)   take((size_t)GG * HH * 4);
    float*    counts= (float*)   take((size_t)GG * 4);

    const int NH = NN * HH;                       // 12,800,000
    const int nh_blocks = (NH + 255) / 256;       // 50,000
    const int n_blocks  = (NN + 255) / 256;
    const int e_blocks  = (EE + 255) / 256;
    const int gemm_blocks = (NN / 16 + 7) / 8;    // 782 (N % 16 == 0: full tiles only)

    // degrees and rsqrt norms
    k_fill1<<<n_blocks, 256, 0, stream>>>(deg, NN);
    k_deg_acc<<<e_blocks, 256, 0, stream>>>(dstv, deg, EE);
    k_rsqrt<<<n_blocks, 256, 0, stream>>>(deg, dinv, NN);

    // layer-0 input in f16
    k_f2h<<<nh_blocks, 256, 0, stream>>>(x, h16, NH);

    for (int l = 0; l < 3; ++l) {
        const float* Wl = (l == 0) ? W0 : (Wrest + (size_t)(l - 1) * HH * HH);
        k_wt<<<(HH * HH + 255) / 256, 256, 0, stream>>>(Wl, Wt);
        k_gemm<<<gemm_blocks, 256, 0, stream>>>(h16, Wt, hW, NN);
        k_agg_init<<<nh_blocks, 256, 0, stream>>>(hW, dinv, bvec + l * HH, agg, NH);
        k_scatter<<<EE / 8, 256, 0, stream>>>(hW, srcv, dstv, dinv, agg);
        k_zero<<<1, 256, 0, stream>>>(stats, 2 * HH);
        k_bn_stats<<<256, 128, 0, stream>>>(agg, stats);
        k_bn_fin<<<1, 128, 0, stream>>>(stats, gamma + l * HH, beta + l * HH, ss);
        k_bn_apply<<<nh_blocks, 256, 0, stream>>>(agg, ss, h16, NH);
    }

    // global mean pool
    k_zero<<<(GG * HH + 255) / 256, 256, 0, stream>>>(pooled, GG * HH);
    k_zero<<<(GG + 255) / 256, 256, 0, stream>>>(counts, GG);
    k_pool<<<nh_blocks, 256, 0, stream>>>(agg, batch, pooled, counts, NH);

    // prediction heads
    k_heads<<<(KK * GG + 255) / 256, 256, 0, stream>>>(pooled, counts, HW1, Hb1, HW2, Hb2, out);
}
